// MoE_77884936946280
// MI455X (gfx1250) — compile-verified, matrix operand order checked by reading, not run on hardware
//
#include <hip/hip_runtime.h>

// ---- problem constants (match reference) ----
#define T_TOK 4096
#define DIM   512
#define NEXP  64
#define HEXP  256
#define HSH   2048
#define LN_EPS 1e-5f
#define MB (1024ull * 1024ull)

typedef __attribute__((ext_vector_type(16))) _Float16 v16h;
typedef __attribute__((ext_vector_type(8)))  _Float16 v8h;
typedef __attribute__((ext_vector_type(8)))  float    v8f;
typedef int v4i_t __attribute__((vector_size(16)));

#if defined(__gfx1250__) && __has_builtin(__builtin_amdgcn_global_load_async_to_lds_b128)
#define HAVE_ASYNC_LDS 1
#endif

__device__ __forceinline__ void wait_asynccnt0() {
#if defined(__gfx1250__)
#if __has_builtin(__builtin_amdgcn_s_wait_asynccnt)
  __builtin_amdgcn_s_wait_asynccnt(0);
#else
  asm volatile("s_wait_asynccnt 0x0" ::: "memory");
#endif
#endif
}

__device__ __forceinline__ float siluf(float v) { return v / (1.0f + __expf(-v)); }

// 16-bit 16x32 fragment from row-major f16 memory (global or LDS), rows = M (A) or N (B^T).
// ISA 7.12.2: lane<16 -> row=lane, k = {0..7, 16..23}; lane>=16 -> k += 8.
// Per lane: two contiguous 16B chunks -> 2x (global|ds)_load_b128, zero conversion.
__device__ __forceinline__ v16h load_frag16(const _Float16* __restrict__ P, int ldk,
                                            int kbase, int row0, int lane) {
  const int hi = lane >> 4, r = (lane & 15) + row0;
  const _Float16* p = P + (size_t)r * ldk + kbase + (hi ? 8 : 0);
  const v8h c0 = *(const v8h*)(p);
  const v8h c1 = *(const v8h*)(p + 16);
  v16h f;
#pragma unroll
  for (int j = 0; j < 8; ++j) { f[j] = c0[j]; f[8 + j] = c1[j]; }
  return f;
}

#define WMMA_F16(A, B, C) \
  __builtin_amdgcn_wmma_f32_16x16x32_f16(false, (A), false, (B), (short)0, (C), false, false)

// ---------------------------------------------------------------------------
__global__ void zero_counts_kernel(int* counts) { counts[threadIdx.x] = 0; }

// Tiled transpose + f32->f16 convert: src[K][N] (f32) -> dst[N][K] (f16).
// blockIdx.z batches over experts (stride K*N).
__global__ __launch_bounds__(256)
void transpose_cvt_kernel(const float* __restrict__ src, _Float16* __restrict__ dst,
                          int K, int N) {
  __shared__ float tile[32][33];
  const float*  s = src + (size_t)blockIdx.z * K * N;
  _Float16*     d = dst + (size_t)blockIdx.z * K * N;
  const int k0 = blockIdx.x * 32, n0 = blockIdx.y * 32;
  const int tx = threadIdx.x & 31, ty = threadIdx.x >> 5;  // 32 x 8
#pragma unroll
  for (int r = 0; r < 32; r += 8)
    tile[ty + r][tx] = s[(size_t)(k0 + ty + r) * N + n0 + tx];
  __syncthreads();
#pragma unroll
  for (int r = 0; r < 32; r += 8)
    d[(size_t)(n0 + ty + r) * K + k0 + tx] = (_Float16)tile[tx][ty + r];
}

// LayerNorm stats; writes nx (f32, for routed per-expert affine) and
// nxs16 = f16(nx*gs+bs) (shared-expert A operand); sigmoid gate; top-2 routing.
__global__ __launch_bounds__(256)
void ln_gate_kernel(const float* __restrict__ x, const float* __restrict__ Wg,
                    const float* __restrict__ bg, const float* __restrict__ gs,
                    const float* __restrict__ bs, float* __restrict__ nx,
                    _Float16* __restrict__ nxs16, int* __restrict__ counts,
                    int* __restrict__ tok_list, float* __restrict__ w_list) {
  __shared__ float xs[DIM];
  __shared__ float red[256];
  __shared__ float scores[NEXP];
  const int t = blockIdx.x;
  const int tid = threadIdx.x;
  const float* xr = x + (size_t)t * DIM;
  const float v0 = xr[tid], v1 = xr[tid + 256];
  xs[tid] = v0; xs[tid + 256] = v1;

  red[tid] = v0 + v1;
  __syncthreads();
  for (int s = 128; s > 0; s >>= 1) { if (tid < s) red[tid] += red[tid + s]; __syncthreads(); }
  const float mu = red[0] * (1.0f / DIM);
  __syncthreads();
  red[tid] = (v0 - mu) * (v0 - mu) + (v1 - mu) * (v1 - mu);
  __syncthreads();
  for (int s = 128; s > 0; s >>= 1) { if (tid < s) red[tid] += red[tid + s]; __syncthreads(); }
  const float rstd = rsqrtf(red[0] * (1.0f / DIM) + LN_EPS);
  const float n0v = (v0 - mu) * rstd, n1v = (v1 - mu) * rstd;
  nx[(size_t)t * DIM + tid]       = n0v;
  nx[(size_t)t * DIM + tid + 256] = n1v;
  nxs16[(size_t)t * DIM + tid]       = (_Float16)(n0v * gs[tid] + bs[tid]);
  nxs16[(size_t)t * DIM + tid + 256] = (_Float16)(n1v * gs[tid + 256] + bs[tid + 256]);

  // gate on raw x: 4 threads per expert, 128 dims each
  const int e = tid >> 2, part = tid & 3;
  float acc = 0.0f;
  const float* wge = Wg + (size_t)e * DIM;
  for (int d = part * 128; d < part * 128 + 128; ++d) acc += xs[d] * wge[d];
  __syncthreads();
  red[tid] = acc;
  __syncthreads();
  if (part == 0) {
    const float s = red[tid] + red[tid + 1] + red[tid + 2] + red[tid + 3] + bg[e];
    scores[e] = 1.0f / (1.0f + __expf(-s));
  }
  __syncthreads();
  if (tid == 0) {
    int i0 = 0; float s0 = -1.0f;
    for (int i = 0; i < NEXP; ++i) if (scores[i] > s0) { s0 = scores[i]; i0 = i; }
    int i1 = (i0 == 0) ? 1 : 0; float s1 = -1.0f;
    for (int i = 0; i < NEXP; ++i) if (i != i0 && scores[i] > s1) { s1 = scores[i]; i1 = i; }
    const float inv = 1.0f / (s0 + s1);
    const int slot0 = atomicAdd(&counts[i0], 1);
    tok_list[i0 * T_TOK + slot0] = t;  w_list[i0 * T_TOK + slot0] = s0 * inv;
    const int slot1 = atomicAdd(&counts[i1], 1);
    tok_list[i1 * T_TOK + slot1] = t;  w_list[i1 * T_TOK + slot1] = s1 * inv;
  }
}

// Shared expert up-proj: hs[T][HS] = silu(nxs@Ws1+bs1) * (nxs@Ws3+bs3), f16 out.
// No LDS: A-fragments come straight from nxs16 (1KB tile, WGP$-resident across waves).
// Block: 16 tokens x 256 cols; 8 waves x 2 n-subtiles; 4 WMMA per A-fragment.
__global__ __launch_bounds__(256)
void shared_h_kernel(const _Float16* __restrict__ nxs16, const _Float16* __restrict__ Ws1t,
                     const float* __restrict__ bs1, const _Float16* __restrict__ Ws3t,
                     const float* __restrict__ bs3, _Float16* __restrict__ hs) {
  const int lane = threadIdx.x & 31, wid = threadIdx.x >> 5;
  const int t0 = blockIdx.x * 16;
  const int nbase = blockIdx.y * 256 + wid * 32;
  v8f acc1[2] = {};
  v8f acc3[2] = {};
  for (int k0 = 0; k0 < DIM; k0 += 32) {
    const v16h a = load_frag16(nxs16, DIM, k0, t0, lane);
#pragma unroll
    for (int s = 0; s < 2; ++s) {
      const v16h b1 = load_frag16(Ws1t, DIM, k0, nbase + s * 16, lane);
      const v16h b3 = load_frag16(Ws3t, DIM, k0, nbase + s * 16, lane);
      acc1[s] = WMMA_F16(a, b1, acc1[s]);
      acc3[s] = WMMA_F16(a, b3, acc3[s]);
    }
  }
  const int hi = lane >> 4;
#pragma unroll
  for (int s = 0; s < 2; ++s) {
    const int nn = (lane & 15) + nbase + s * 16;
#pragma unroll
    for (int r = 0; r < 8; ++r) {
      const int m = r + hi * 8;
      const float h1 = acc1[s][r] + bs1[nn];
      const float h3 = acc3[s][r] + bs3[nn];
      hs[(size_t)(t0 + m) * HSH + nn] = (_Float16)(siluf(h1) * h3);
    }
  }
}

// Shared expert down-proj: out[T][D] = hs @ Ws2 + bs2 (plain stores init out).
// Block: 16 tokens x 256 cols; 8 waves x 2 n-subtiles; A direct from hs (f16).
__global__ __launch_bounds__(256)
void shared_out_kernel(const _Float16* __restrict__ hs, const _Float16* __restrict__ Ws2t,
                       const float* __restrict__ bs2, float* __restrict__ out) {
  const int lane = threadIdx.x & 31, wid = threadIdx.x >> 5;
  const int t0 = blockIdx.x * 16;
  const int nbase = blockIdx.y * 256 + wid * 32;
  v8f acc[2] = {};
  for (int k0 = 0; k0 < HSH; k0 += 32) {
    const v16h a = load_frag16(hs, HSH, k0, t0, lane);
#pragma unroll
    for (int s = 0; s < 2; ++s) {
      const v16h b = load_frag16(Ws2t, HSH, k0, nbase + s * 16, lane);
      acc[s] = WMMA_F16(a, b, acc[s]);
    }
  }
  const int hi = lane >> 4;
#pragma unroll
  for (int s = 0; s < 2; ++s) {
    const int nn = (lane & 15) + nbase + s * 16;
#pragma unroll
    for (int r = 0; r < 8; ++r) {
      const int m = r + hi * 8;
      out[(size_t)(t0 + m) * DIM + nn] = acc[s][r] + bs2[nn];
    }
  }
}

// Routed experts: grid (expert, tile-slice); async-gather 16-token tiles into LDS,
// apply per-expert LN affine, then two WMMA GEMMs.
#define RT_SLICES 8
__global__ __launch_bounds__(256)
void routed_kernel(const float* __restrict__ nx, const float* __restrict__ gamma,
                   const float* __restrict__ beta,
                   const _Float16* __restrict__ W1t, const float* __restrict__ b1,
                   const _Float16* __restrict__ W3t, const float* __restrict__ b3,
                   const _Float16* __restrict__ W2t, const float* __restrict__ b2,
                   const int* __restrict__ counts, const int* __restrict__ tok_list,
                   const float* __restrict__ w_list, float* __restrict__ out) {
#if defined(HAVE_ASYNC_LDS)
  __shared__ float Xs[16 * DIM];      // 32 KB: raw gathered token rows (async landing zone)
#endif
  __shared__ _Float16 As[16 * DIM];   // 16 KB: LN-affine'd token tile (f16)
  __shared__ _Float16 Hs[16 * HEXP];  //  8 KB: routing-weighted SwiGLU tile
  __shared__ int   tids[16];
  __shared__ float wts[16];
  const int e = blockIdx.x;
  const int tid = threadIdx.x, lane = tid & 31, wid = tid >> 5;
  const int nt = counts[e];
  const float* g   = gamma + (size_t)e * DIM;
  const float* bt  = beta  + (size_t)e * DIM;
  const _Float16* W1e = W1t + (size_t)e * DIM * HEXP;   // [HEXP][DIM]
  const _Float16* W3e = W3t + (size_t)e * DIM * HEXP;   // [HEXP][DIM]
  const _Float16* W2e = W2t + (size_t)e * HEXP * DIM;   // [DIM][HEXP]

  for (int tile = blockIdx.y * 16; tile < nt; tile += RT_SLICES * 16) {
    if (tid < 16) {
      const int slot = tile + tid;
      tids[tid] = (slot < nt) ? tok_list[e * T_TOK + slot] : -1;
      wts[tid]  = (slot < nt) ? w_list[e * T_TOK + slot]  : 0.0f;
    }
    __syncthreads();

#if defined(HAVE_ASYNC_LDS)
    // async DMA the 16 gathered rows (f32) into LDS: 2048 x 16B chunks, 8 per thread
#pragma unroll
    for (int cc = 0; cc < 8; ++cc) {
      const int c = cc * 256 + tid;
      const int m = c >> 7, q = c & 127;          // row, 16B-chunk within row
      const int t = tids[m];
      if (t >= 0) {
        __builtin_amdgcn_global_load_async_to_lds_b128(
            (v4i_t*)(nx + (size_t)t * DIM + q * 4),
            (v4i_t*)&Xs[m * DIM + q * 4], 0, 0);
      }
    }
    wait_asynccnt0();
    __syncthreads();
    for (int i = tid; i < 16 * DIM; i += 256) {
      const int m = i / DIM, k = i % DIM;
      As[i] = (tids[m] >= 0) ? (_Float16)(Xs[i] * g[k] + bt[k]) : (_Float16)0.0f;
    }
#else
    for (int i = tid; i < 16 * DIM; i += 256) {
      const int m = i / DIM, k = i % DIM;
      const int t = tids[m];
      const float v = (t >= 0) ? (nx[(size_t)t * DIM + k] * g[k] + bt[k]) : 0.0f;
      As[i] = (_Float16)v;
    }
#endif
    __syncthreads();

    // phase 1: Hs = w * ( silu(A@W1+b1) * (A@W3+b3) ); 8 waves x 2 n-tiles = 256 cols
#pragma unroll 1
    for (int sub = 0; sub < 2; ++sub) {
      const int n0 = (wid * 2 + sub) * 16;
      v8f a1 = {}; v8f a3 = {};
      for (int k0 = 0; k0 < DIM; k0 += 32) {
        const v16h a   = load_frag16(As, DIM, k0, 0, lane);
        const v16h bb1 = load_frag16(W1e, DIM, k0, n0, lane);
        const v16h bb3 = load_frag16(W3e, DIM, k0, n0, lane);
        if (k0 + 32 < DIM)
          __builtin_prefetch(W1e + (size_t)(n0 + (lane & 15)) * DIM + k0 + 32, 0, 3);
        a1 = WMMA_F16(a, bb1, a1);
        a3 = WMMA_F16(a, bb3, a3);
      }
      const int hi = lane >> 4, nn = (lane & 15) + n0;
#pragma unroll
      for (int r = 0; r < 8; ++r) {
        const int m = r + hi * 8;
        const float h1 = a1[r] + b1[e * HEXP + nn];
        const float h3 = a3[r] + b3[e * HEXP + nn];
        Hs[m * HEXP + nn] = (_Float16)(siluf(h1) * h3 * wts[m]);
      }
    }
    __syncthreads();

    // phase 2: out += Hs @ W2 + w*b2; 8 waves x 4 n-tiles = 512 cols
#pragma unroll 1
    for (int sub = 0; sub < 4; ++sub) {
      const int n0 = (wid * 4 + sub) * 16;
      v8f acc = {};
      for (int k0 = 0; k0 < HEXP; k0 += 32) {
        const v16h a  = load_frag16(Hs, HEXP, k0, 0, lane);
        const v16h bb = load_frag16(W2e, HEXP, k0, n0, lane);
        acc = WMMA_F16(a, bb, acc);
      }
      const int hi = lane >> 4, nn = (lane & 15) + n0;
#pragma unroll
      for (int r = 0; r < 8; ++r) {
        const int m = r + hi * 8;
        const int t = tids[m];
        if (t >= 0)
          atomicAdd(&out[(size_t)t * DIM + nn], acc[r] + wts[m] * b2[e * DIM + nn]);
      }
    }
    __syncthreads();
  }
}

// ---------------------------------------------------------------------------
extern "C" void kernel_launch(void* const* d_in, const int* in_sizes, int n_in,
                              void* d_out, int out_size, void* d_ws, size_t ws_size,
                              hipStream_t stream) {
  const float* x     = (const float*)d_in[0];
  const float* Wg    = (const float*)d_in[1];
  const float* bg    = (const float*)d_in[2];
  const float* gamma = (const float*)d_in[3];
  const float* beta  = (const float*)d_in[4];
  const float* W1    = (const float*)d_in[5];
  const float* b1    = (const float*)d_in[6];
  const float* W3    = (const float*)d_in[7];
  const float* b3    = (const float*)d_in[8];
  const float* W2    = (const float*)d_in[9];
  const float* b2    = (const float*)d_in[10];
  const float* gs    = (const float*)d_in[11];
  const float* bs    = (const float*)d_in[12];
  const float* Ws1   = (const float*)d_in[13];
  const float* bs1   = (const float*)d_in[14];
  const float* Ws3   = (const float*)d_in[15];
  const float* bs3   = (const float*)d_in[16];
  const float* Ws2   = (const float*)d_in[17];
  const float* bs2   = (const float*)d_in[18];
  float* out = (float*)d_out;

  // workspace layout
  char* ws = (char*)d_ws;
  float*    nx       = (float*)    ws;                       //  8 MB
  _Float16* nxs16    = (_Float16*)(ws +  8 * MB);            //  4 MB
  _Float16* hs       = (_Float16*)(ws + 12 * MB);            // 16 MB
  int*      counts   = (int*)     (ws + 28 * MB);            // 256 B
  int*      tok_list = (int*)     (ws + 28 * MB + 4096);     //  1 MB
  float*    w_list   = (float*)   (ws + 29 * MB + 4096);     //  1 MB
  _Float16* Ws1t     = (_Float16*)(ws + 31 * MB);            //  2 MB  [HS][D]
  _Float16* Ws3t     = (_Float16*)(ws + 33 * MB);            //  2 MB  [HS][D]
  _Float16* Ws2t     = (_Float16*)(ws + 35 * MB);            //  2 MB  [D][HS]
  _Float16* W1t      = (_Float16*)(ws + 37 * MB);            // 16 MB  E x [H][D]
  _Float16* W3t      = (_Float16*)(ws + 53 * MB);            // 16 MB  E x [H][D]
  _Float16* W2t      = (_Float16*)(ws + 69 * MB);            // 16 MB  E x [D][H]

  // one-time (per launch) weight transpose + f16 convert
  transpose_cvt_kernel<<<dim3(DIM / 32, HSH / 32, 1), 256, 0, stream>>>(Ws1, Ws1t, DIM, HSH);
  transpose_cvt_kernel<<<dim3(DIM / 32, HSH / 32, 1), 256, 0, stream>>>(Ws3, Ws3t, DIM, HSH);
  transpose_cvt_kernel<<<dim3(HSH / 32, DIM / 32, 1), 256, 0, stream>>>(Ws2, Ws2t, HSH, DIM);
  transpose_cvt_kernel<<<dim3(DIM / 32, HEXP / 32, NEXP), 256, 0, stream>>>(W1, W1t, DIM, HEXP);
  transpose_cvt_kernel<<<dim3(DIM / 32, HEXP / 32, NEXP), 256, 0, stream>>>(W3, W3t, DIM, HEXP);
  transpose_cvt_kernel<<<dim3(HEXP / 32, DIM / 32, NEXP), 256, 0, stream>>>(W2, W2t, HEXP, DIM);

  zero_counts_kernel<<<1, NEXP, 0, stream>>>(counts);
  ln_gate_kernel<<<T_TOK, 256, 0, stream>>>(x, Wg, bg, gs, bs, nx, nxs16,
                                            counts, tok_list, w_list);
  shared_h_kernel<<<dim3(T_TOK / 16, HSH / 256), 256, 0, stream>>>(nxs16, Ws1t, bs1,
                                                                   Ws3t, bs3, hs);
  shared_out_kernel<<<dim3(T_TOK / 16, DIM / 256), 256, 0, stream>>>(hs, Ws2t, bs2, out);
  routed_kernel<<<dim3(NEXP, RT_SLICES), 256, 0, stream>>>(nx, gamma, beta, W1t, b1, W3t, b3,
                                                           W2t, b2, counts, tok_list, w_list, out);
}
